// CrystalAttention_38740605009994
// MI455X (gfx1250) — matrix operand order repaired and checked by reading, not compiled, not run on hardware
//
#include <hip/hip_runtime.h>
#include <stdint.h>

typedef __attribute__((ext_vector_type(16))) __bf16       v16bf;
typedef __attribute__((ext_vector_type(8)))  float        v8f;
typedef __attribute__((ext_vector_type(4)))  unsigned int v4u;
typedef __attribute__((ext_vector_type(8)))  int          v8i;
typedef __attribute__((ext_vector_type(4)))  int          v4i;

// ---------------- workspace layout (bytes) ----------------
static constexpr size_t WS_POS  = 0;                                  // 1024*512 bf16 = 1 MB
static constexpr size_t WS_VALT = (size_t)1024 * 512 * 2;             // valT[d*1024+n], 1 MB
static constexpr size_t WS_WBF  = WS_VALT + (size_t)512 * 1024 * 2;   // 512*512 bf16 = 0.5 MB
static constexpr size_t WS_P2   = WS_WBF  + (size_t)512 * 512 * 2;    // 1024 f32

// ---------------- prep: f32 -> bf16 weights, p2 ----------------
__global__ void crystal_prep_kernel(const float* __restrict__ pos,
                                    const float* __restrict__ val,
                                    const float* __restrict__ w,
                                    __bf16* __restrict__ posbf,
                                    __bf16* __restrict__ valT,
                                    __bf16* __restrict__ wbf,
                                    float*  __restrict__ p2)
{
    int tid  = blockIdx.x * blockDim.x + threadIdx.x;
    int nthr = gridDim.x * blockDim.x;
    for (int i = tid; i < 1024 * 512; i += nthr) posbf[i] = (__bf16)pos[i];
    for (int i = tid; i < 1024 * 512; i += nthr) {
        int n = i & 1023, d = i >> 10;                 // i = d*1024 + n
        valT[i] = (__bf16)val[n * 512 + d];
    }
    for (int i = tid; i < 512 * 512; i += nthr) wbf[i] = (__bf16)w[i];
    for (int n = tid; n < 1024; n += nthr) {
        float s = 0.f;
        for (int d = 0; d < 512; ++d) { float v = pos[n * 512 + d]; s += v * v; }
        p2[n] = s;
    }
}

// ---------------- fragment builders (ISA 7.12.2, 16-bit layouts) ----------------
__device__ __forceinline__ v16bf load_a_frag(const __bf16* __restrict__ row, int k0, int half)
{
    v16bf a;
    int base = k0 + (half ? 8 : 0);
#pragma unroll
    for (int v = 0; v < 8; ++v) {
        int kk = base + ((v < 4) ? (2 * v) : (16 + 2 * (v - 4)));
        a[2 * v]     = row[kk];
        a[2 * v + 1] = row[kk + 1];
    }
    return a;
}
__device__ __forceinline__ v16bf load_b_frag(const __bf16* __restrict__ col_k, int half)
{
    v16bf b;
    const __bf16* p = col_k + (half ? 8 : 0);
#pragma unroll
    for (int e = 0; e < 16; ++e) b[e] = p[(e < 8) ? e : (e + 8)];
    return b;
}
__device__ __forceinline__ v8f zero8()
{
    v8f z;
#pragma unroll
    for (int r = 0; r < 8; ++r) z[r] = 0.f;
    return z;
}
__device__ __forceinline__ v8f wmma_bf16(v16bf a, v16bf b, v8f c)
{
    return __builtin_amdgcn_wmma_f32_16x16x32_bf16(false, a, false, b, (short)0, c, false, false);
}

// TDM descriptor for a 32-row x 512-col f32 tile of a 16384x512 row-major tensor.
// (ISA 8.3/8.4: group0 = count|lds|global|type, group1 = data_size/dims/strides)
__device__ __forceinline__ void tdm_desc_32x512_f32(v4u& g0, v8i& g1,
                                                    uint64_t ga, uint32_t lds_off)
{
    g0[0] = 1u;                                             // count=1, user mode
    g0[1] = lds_off;                                        // lds_addr
    g0[2] = (uint32_t)ga;                                   // global_addr[31:0]
    g0[3] = (uint32_t)((ga >> 32) & 0x1FFFFFFu) | (2u << 30); // addr[56:32] | type=2
    g1[0] = (int)0x00020000;                                // data_size=2 -> 4 bytes
    g1[1] = (int)(512u << 16);                              // tensor_dim0 = 512
    g1[2] = (int)(16384u << 16);                            // tensor_dim1 = 16384
    g1[3] = (int)(512u << 16);                              // tile_dim0 = 512
    g1[4] = 32;                                             // tile_dim1 = 32, tile_dim2 = 0
    g1[5] = 512;                                            // tensor_dim0_stride = 512
    g1[6] = 0;
    g1[7] = 0;
}

// ---------------- LDS layout (~97.4 KB arena, phase-overlaid) ----------------
// [0, 64K)   : ph0-1: x tile f32 (TDM dest) | ph2-3: P = exp bf16 | ph4: Ofin f32 (TDM src)
// [64K, 96K) : ph1-2: xbf (32x512 bf16)     | ph3+: Obf (32x512 bf16)
// [96K, ...] : red[32][8], x2s[32], rowsum[32], rowinv[32]
#define SM_XB  (64 * 1024)
#define SM_RED (96 * 1024)
#define SM_X2  (SM_RED + 32 * 8 * 4)
#define SM_RS  (SM_X2 + 32 * 4)
#define SM_RI  (SM_RS + 32 * 4)
#define SM_TOT (SM_RI + 32 * 4)

#define EXP_SHIFT 25.0f

// ---------------- main fused kernel: 32 rows per WG, 8 waves ----------------
__global__ __launch_bounds__(256)
void crystal_main_kernel(const float* __restrict__ x,
                         const float* __restrict__ scales,
                         const float* __restrict__ b_out,
                         const __bf16* __restrict__ posbf,
                         const __bf16* __restrict__ valT,
                         const __bf16* __restrict__ wbf,
                         const float*  __restrict__ p2,
                         float* __restrict__ out)
{
    __shared__ __align__(16) unsigned char smem[SM_TOT];
    float*  xf32   = (float*)smem;                    // 32x512 f32 (TDM staging in)
    __bf16* P      = (__bf16*)smem;                   // 32x1024 bf16 (overlay)
    float*  Ofin   = (float*)smem;                    // 32x512 f32 (overlay, TDM staging out)
    __bf16* xbf    = (__bf16*)(smem + SM_XB);         // 32x512 bf16
    __bf16* Obf    = (__bf16*)(smem + SM_XB);         // overlay after phase 2
    float (*red)[8]= (float (*)[8])(smem + SM_RED);
    float*  x2s    = (float*)(smem + SM_X2);
    float*  rowsum = (float*)(smem + SM_RS);
    float*  rowinv = (float*)(smem + SM_RI);

    const int tid    = threadIdx.x;
    const int wave   = tid >> 5;
    const int lane   = tid & 31;
    const int half   = lane >> 4;
    const int lane16 = lane & 15;
    const int row0   = blockIdx.x * 32;   // 512 blocks cover 16384 rows

    // ---- Phase 0: TDM async copy of the 32x512 f32 x-tile into LDS ----
#if __has_builtin(__builtin_amdgcn_tensor_load_to_lds)
    if (wave == 0) {
        v4u g0; v8i g1;
        tdm_desc_32x512_f32(g0, g1,
                            (uint64_t)(uintptr_t)(x + (size_t)row0 * 512),
                            (uint32_t)(uintptr_t)xf32);
        v4i gz; gz[0] = 0; gz[1] = 0; gz[2] = 0; gz[3] = 0;
#if defined(__clang_major__) && (__clang_major__ >= 23)
        v8i gz8; for (int i = 0; i < 8; ++i) gz8[i] = 0;
        __builtin_amdgcn_tensor_load_to_lds(g0, g1, gz, gz, gz8, 0);
#else
        __builtin_amdgcn_tensor_load_to_lds(g0, g1, gz, gz, 0);
#endif
#if __has_builtin(__builtin_amdgcn_s_wait_tensorcnt)
        __builtin_amdgcn_s_wait_tensorcnt(0);
#endif
    }
    __syncthreads();
    const bool x_in_lds = true;
#else
    const bool x_in_lds = false;
#endif

    // ---- Phase 1: f32 -> bf16 x tile + row sum of squares; zero row sums ----
    {
        int m = tid >> 3, sub = tid & 7;          // 32 rows x 8 chunks of 64
        const float* xr = x_in_lds ? (xf32 + m * 512 + sub * 64)
                                   : (x + (size_t)(row0 + m) * 512 + sub * 64);
        float ss = 0.f;
#pragma unroll 8
        for (int c = 0; c < 64; ++c) {
            float v = xr[c];
            ss += v * v;
            xbf[m * 512 + sub * 64 + c] = (__bf16)v;
        }
        red[m][sub] = ss;
        if (tid < 32) rowsum[tid] = 0.f;
    }
    __syncthreads();
    if (tid < 32) {
        float s = 0.f;
#pragma unroll
        for (int i = 0; i < 8; ++i) s += red[tid][i];
        x2s[tid] = s;
    }
    __syncthreads();

    // ---- Phase 2: logits = f(x @ pos^T); write P = exp(logit-25) bf16; row sums ----
    {
        float rs0[8], rs1[8];
#pragma unroll
        for (int r = 0; r < 8; ++r) { rs0[r] = 0.f; rs1[r] = 0.f; }
        for (int nt = wave; nt < 64; nt += 8) {
            const int n0 = nt * 16;
            v8f acc0 = zero8(), acc1 = zero8();
            const __bf16* prow = posbf + (size_t)(n0 + lane16) * 512;
            for (int k = 0; k < 512; k += 32) {
                v16bf b  = load_b_frag(prow + k, half);
                v16bf a0 = load_a_frag(xbf + lane16 * 512, k, half);
                v16bf a1 = load_a_frag(xbf + (16 + lane16) * 512, k, half);
                acc0 = wmma_bf16(a0, b, acc0);
                acc1 = wmma_bf16(a1, b, acc1);
            }
            const int   n_abs = n0 + lane16;
            const float p2n   = p2[n_abs];
            const float scn   = scales[n_abs];
#pragma unroll
            for (int r = 0; r < 8; ++r) {
                int m = r + 8 * half;
                float d2 = x2s[m] - 2.0f * acc0[r] + p2n;
                d2 = d2 > 0.f ? d2 : 0.f;
                float e = __expf(scn / (sqrtf(d2) + 0.1f) - EXP_SHIFT);
                rs0[r] += e;
                P[m * 1024 + n_abs] = (__bf16)e;

                int m1 = 16 + m;
                float d2b = x2s[m1] - 2.0f * acc1[r] + p2n;
                d2b = d2b > 0.f ? d2b : 0.f;
                float e1 = __expf(scn / (sqrtf(d2b) + 0.1f) - EXP_SHIFT);
                rs1[r] += e1;
                P[m1 * 1024 + n_abs] = (__bf16)e1;
            }
        }
#pragma unroll
        for (int r = 0; r < 8; ++r) {
            float s0 = rs0[r], s1 = rs1[r];
#pragma unroll
            for (int mk = 1; mk < 16; mk <<= 1) {
                s0 += __shfl_xor(s0, mk);
                s1 += __shfl_xor(s1, mk);
            }
            if (lane16 == 0) {
                atomicAdd(&rowsum[r + 8 * half], s0);
                atomicAdd(&rowsum[16 + r + 8 * half], s1);
            }
        }
    }
    __syncthreads();
    if (tid < 32) rowinv[tid] = 1.0f / rowsum[tid];
    __syncthreads();

    // ---- Phase 3: O = P @ V (K=1024); scale by 1/rowsum; write Obf (overlays xbf) ----
    {
        v8f acc0[4], acc1[4];
#pragma unroll
        for (int jt = 0; jt < 4; ++jt) { acc0[jt] = zero8(); acc1[jt] = zero8(); }
        const int d0 = wave * 64;
        for (int k = 0; k < 1024; k += 32) {
            v16bf a0 = load_a_frag(P + lane16 * 1024, k, half);
            v16bf a1 = load_a_frag(P + (16 + lane16) * 1024, k, half);
#pragma unroll
            for (int jt = 0; jt < 4; ++jt) {
                int d = d0 + jt * 16 + lane16;
                v16bf b = load_b_frag(valT + (size_t)d * 1024 + k, half);
                acc0[jt] = wmma_bf16(a0, b, acc0[jt]);
                acc1[jt] = wmma_bf16(a1, b, acc1[jt]);
            }
        }
        __syncthreads();   // everyone done reading P before Ofin overlays it (phase 4)
#pragma unroll
        for (int jt = 0; jt < 4; ++jt) {
            int d = d0 + jt * 16 + lane16;
#pragma unroll
            for (int r = 0; r < 8; ++r) {
                int m = r + 8 * half;
                Obf[m * 512 + d]        = (__bf16)(acc0[jt][r] * rowinv[m]);
                Obf[(16 + m) * 512 + d] = (__bf16)(acc1[jt][r] * rowinv[16 + m]);
            }
        }
    }
    __syncthreads();

    // ---- Phase 4: final = O @ w_out^T + b_out; stage f32 tile in LDS; TDM store ----
    {
        const int j0 = wave * 64;
        v8f acc0[4], acc1[4];
#pragma unroll
        for (int jt = 0; jt < 4; ++jt) {
            float bj = b_out[j0 + jt * 16 + lane16];
            v8f c;
#pragma unroll
            for (int r = 0; r < 8; ++r) c[r] = bj;
            acc0[jt] = c; acc1[jt] = c;
        }
        for (int k = 0; k < 512; k += 32) {
            v16bf a0 = load_a_frag(Obf + lane16 * 512, k, half);
            v16bf a1 = load_a_frag(Obf + (16 + lane16) * 512, k, half);
#pragma unroll
            for (int jt = 0; jt < 4; ++jt) {
                int j = j0 + jt * 16 + lane16;
                v16bf b = load_b_frag(wbf + (size_t)j * 512 + k, half);
                acc0[jt] = wmma_bf16(a0, b, acc0[jt]);
                acc1[jt] = wmma_bf16(a1, b, acc1[jt]);
            }
        }
#if __has_builtin(__builtin_amdgcn_tensor_store_from_lds)
        // stage into LDS (overlays dead P region), then one coalesced TDM store
#pragma unroll
        for (int jt = 0; jt < 4; ++jt) {
            int j = j0 + jt * 16 + lane16;
#pragma unroll
            for (int r = 0; r < 8; ++r) {
                int m = r + 8 * half;
                Ofin[m * 512 + j]        = acc0[jt][r];
                Ofin[(16 + m) * 512 + j] = acc1[jt][r];
            }
        }
        __syncthreads();
        if (wave == 0) {
            v4u g0; v8i g1;
            tdm_desc_32x512_f32(g0, g1,
                                (uint64_t)(uintptr_t)(out + (size_t)row0 * 512),
                                (uint32_t)(uintptr_t)Ofin);
            v4i gz; gz[0] = 0; gz[1] = 0; gz[2] = 0; gz[3] = 0;
#if defined(__clang_major__) && (__clang_major__ >= 23)
            v8i gz8; for (int i = 0; i < 8; ++i) gz8[i] = 0;
            __builtin_amdgcn_tensor_store_from_lds(g0, g1, gz, gz, gz8, 0);
#else
            __builtin_amdgcn_tensor_store_from_lds(g0, g1, gz, gz, 0);
#endif
#if __has_builtin(__builtin_amdgcn_s_wait_tensorcnt)
            __builtin_amdgcn_s_wait_tensorcnt(0);
#endif
        }
#else
        // fallback: direct (column-strided) stores
#pragma unroll
        for (int jt = 0; jt < 4; ++jt) {
            int j = j0 + jt * 16 + lane16;
#pragma unroll
            for (int r = 0; r < 8; ++r) {
                int m = r + 8 * half;
                out[(size_t)(row0 + m) * 512 + j]      = acc0[jt][r];
                out[(size_t)(row0 + 16 + m) * 512 + j] = acc1[jt][r];
            }
        }
#endif
    }
}

// ---------------- launch ----------------
extern "C" void kernel_launch(void* const* d_in, const int* in_sizes, int n_in,
                              void* d_out, int out_size, void* d_ws, size_t ws_size,
                              hipStream_t stream)
{
    const float* x   = (const float*)d_in[0];
    const float* pos = (const float*)d_in[1];
    const float* sc  = (const float*)d_in[2];
    const float* val = (const float*)d_in[3];
    const float* w   = (const float*)d_in[4];
    const float* b   = (const float*)d_in[5];

    char* ws = (char*)d_ws;
    __bf16* posbf = (__bf16*)(ws + WS_POS);
    __bf16* valT  = (__bf16*)(ws + WS_VALT);
    __bf16* wbf   = (__bf16*)(ws + WS_WBF);
    float*  p2    = (float*)(ws + WS_P2);

    crystal_prep_kernel<<<512, 256, 0, stream>>>(pos, val, w, posbf, valT, wbf, p2);
    crystal_main_kernel<<<512, 256, 0, stream>>>(x, sc, b, posbf, valT, wbf, p2,
                                                 (float*)d_out);
}
